// M2_52948356825792
// MI455X (gfx1250) — matrix operand order repaired and checked by reading, not compiled, not run on hardware
//
#include <hip/hip_runtime.h>

// ---------------- problem constants ----------------
#define BDIM   2
#define TDIM   2048
#define HDIM   512
#define NHEAD  8
#define HEADD  64
#define VDIM   32000
#define MROWS  (BDIM * TDIM)   // 4096

typedef __attribute__((ext_vector_type(16))) __bf16 v16bf;
typedef __attribute__((ext_vector_type(8)))  __bf16 v8bf;
typedef __attribute__((ext_vector_type(8)))  float  v8f;
typedef __attribute__((ext_vector_type(4)))  unsigned tdm_v4u;
typedef __attribute__((ext_vector_type(8)))  int      tdm_v8i;
typedef __attribute__((ext_vector_type(4)))  int      tdm_v4i;

__device__ __forceinline__ __bf16 f2bf(float f) {
  unsigned u = __builtin_bit_cast(unsigned, f);
  u += 0x7fffu + ((u >> 16) & 1u);              // round-to-nearest-even
  return __builtin_bit_cast(__bf16, (unsigned short)(u >> 16));
}

__device__ __forceinline__ v16bf cat8(v8bf lo, v8bf hi) {
  return __builtin_shufflevector(lo, hi, 0,1,2,3,4,5,6,7,8,9,10,11,12,13,14,15);
}

__device__ __forceinline__ float rmax16(float x) {
#pragma unroll
  for (int m = 8; m >= 1; m >>= 1) x = fmaxf(x, __shfl_xor(x, m, 16));
  return x;
}
__device__ __forceinline__ float rsum16(float x) {
#pragma unroll
  for (int m = 8; m >= 1; m >>= 1) x += __shfl_xor(x, m, 16);
  return x;
}

// low 32 bits of a flat shared-aperture address == LDS byte offset (ISA 10.2)
__device__ __forceinline__ unsigned lds_off(const void* p) {
  return (unsigned)(uintptr_t)p;
}

// CDNA5 async DMA: global -> LDS, tracked by ASYNCcnt (CDNA5 ISA)
__device__ __forceinline__ void async_ld_b128(unsigned ldsa, const void* g) {
  asm volatile("global_load_async_to_lds_b128 %0, %1, off"
               :: "v"(ldsa), "v"((unsigned long long)(uintptr_t)g) : "memory");
}
__device__ __forceinline__ void wait_async0() {
  asm volatile("s_wait_asynccnt 0x0" ::: "memory");
}

// CDNA5 Tensor Data Mover: 2D tile (32 elems wide x `rows`) of bf16 from global
// to LDS with 32B padding every 64B row (-> 96B LDS row stride). D# packing per
// CDNA5 ISA. Tracked by TENSORcnt.
__device__ __forceinline__ void tdm_load_2d_b(unsigned lds, const void* g,
                                              int rows, int rowstride_elems) {
  unsigned long long ga = (unsigned long long)(uintptr_t)g;
  tdm_v4u g0;
  g0[0] = 1u;                                                   // count=1 (valid), user mode
  g0[1] = lds;                                                  // lds_addr
  g0[2] = (unsigned)ga;                                         // global_addr[31:0]
  g0[3] = (unsigned)((ga >> 32) & 0x01FFFFFFull) | (2u << 30);  // global_addr[56:32], type=2
  tdm_v8i g1;
  g1[0] = (int)((1u << 16)    // data_size = 1 -> 2 bytes
              | (1u << 20)    // pad_enable
              | (3u << 22)    // pad_interval: every 16 DWORDs (64B)
              | (7u << 25));  // pad_amount: 8 DWORDs (32B)
  g1[1] = (int)(32u << 16);                                     // tensor_dim0 = 32
  g1[2] = (int)(((unsigned)rows & 0xFFFFu) << 16);              // tensor_dim1 lo16
  g1[3] = (int)((((unsigned)rows >> 16) & 0xFFFFu) | (32u << 16)); // dim1 hi16 | tile_dim0=32
  g1[4] = (int)((unsigned)rows & 0xFFFFu);                      // tile_dim1 = rows, tile_dim2=0
  g1[5] = rowstride_elems;                                      // tensor_dim0_stride lo32
  g1[6] = 0;                                                    // stride hi, dim1_stride lo
  g1[7] = 0;
  tdm_v4i z4 = {};
#if __clang_major__ >= 23
  tdm_v8i z8 = {};
  __builtin_amdgcn_tensor_load_to_lds(g0, g1, z4, z4, z8, 0);
#else
  __builtin_amdgcn_tensor_load_to_lds(g0, g1, z4, z4, 0);
#endif
}

// ---------------- elementwise kernels ----------------
__global__ void k_f32_to_bf16(const float* __restrict__ src, __bf16* __restrict__ dst, int n) {
  int i = blockIdx.x * 256 + threadIdx.x;
  if (i < n) dst[i] = f2bf(src[i]);
}

// xe[b*T+t][h] = bf16(tok_emb[ixs[b,t]][h] + pos_emb[t][h])
__global__ void k_embed(const int* __restrict__ ixs, const float* __restrict__ tok,
                        const float* __restrict__ pos, __bf16* __restrict__ xe) {
  int i = blockIdx.x * 256 + threadIdx.x;        // over MROWS*HDIM
  int h   = i & (HDIM - 1);
  int row = i >> 9;                              // HDIM = 512
  int t   = row & (TDIM - 1);
  int ix  = ixs[row];
  xe[i] = f2bf(tok[(size_t)ix * HDIM + h] + pos[(size_t)t * HDIM + h]);
}

// vt[((b*NH+h)*HD + d)*T + t] = v[(b*T+t)*H + h*HD + d]
__global__ void k_transpose_v(const __bf16* __restrict__ v, __bf16* __restrict__ vt) {
  int i = blockIdx.x * 256 + threadIdx.x;        // over B*NH*HD*T = MROWS*HDIM
  int t  = i & (TDIM - 1);
  int d  = (i >> 11) & (HEADD - 1);
  int bh = i >> 17;
  int b  = bh >> 3, h = bh & 7;
  vt[i] = v[((size_t)(b * TDIM + t)) * HDIM + h * HEADD + d];
}

// ---------------- WMMA NT-GEMM: C[M,N] = A[M,K] * B[N,K]^T (+bias)(+relu) ----------------
// Block: 256 threads = 8 waves, tile 128(M) x 256(N), BK=32. Wave grid 2(M) x 4(N),
// wave tile 64x64 = 4x4 WMMA tiles (16 wmma : 16 ds_load_b128 per K-step).
// Double-buffered LDS. A tile (8KB) filled by per-lane GLOBAL_LOAD_ASYNC_TO_LDS_B128
// (2 per thread, ASYNCcnt); B tile (16KB) filled by one TENSOR_LOAD_TO_LDS descriptor
// issued by wave 0 (TENSORcnt). Pipeline: wait(cur) -> barrier -> issue(next) -> compute.
template <int RELU, int BIAS, int OUTBF>
__global__ __launch_bounds__(256) void k_gemm_nt(
    const __bf16* __restrict__ A, const __bf16* __restrict__ Bm,
    const float* __restrict__ bias, float* __restrict__ Cf, __bf16* __restrict__ Cb,
    int Mdim, int Ndim, int Kdim)
{
  __shared__ __bf16 As[2][128][48];   // 2 x 12 KB
  __shared__ __bf16 Bs[2][256][48];   // 2 x 24 KB

  const int tid  = threadIdx.x;
  const int lane = tid & 31;
  const int w    = tid >> 5;
  const int wm   = w & 1;          // 0..1
  const int wn   = w >> 1;         // 0..3
  const int m0   = blockIdx.y * 128;
  const int n0   = blockIdx.x * 256;

  // A cooperative async-load mapping: 128 rows x 32 halfs = 512 b128 -> 2/thread
  const int arow = tid >> 1;
  const int acol = (tid & 1) * 16;
  const __bf16* Ag = A  + (size_t)(m0 + arow) * Kdim + acol;
  const __bf16* Bg = Bm + (size_t)n0 * Kdim;      // B tile base (TDM steps rows)

  const int ln  = lane & 15;
  const int g   = lane >> 4;
  const int kbA = g * 8;            // A-frag K-chunk base (16-bit A 16x32 layout)
  const int kbB = g * 8;            // B-frag K-chunk base (same order as A)

  v8f acc[4][4] = {};

  auto issue = [&](int buf, int k0) {
    unsigned d = lds_off(&As[buf][arow][acol]);
    async_ld_b128(d,      Ag + k0);
    async_ld_b128(d + 16, Ag + k0 + 8);
    if (w == 0)
      tdm_load_2d_b(lds_off(&Bs[buf][0][0]), Bg + k0, 256, Kdim);
  };

  issue(0, 0);

  int cur = 0;
  for (int k0 = 0; k0 < Kdim; k0 += 32, cur ^= 1) {
    wait_async0();                          // this wave's A-tile DMA done (only cur in flight)
    __builtin_amdgcn_s_wait_tensorcnt(0);   // wave0: B-tile TDM done (others: no-op)
    __syncthreads();                        // publish cur; all prev-buffer reads retired
    if (k0 + 32 < Kdim) issue(cur ^ 1, k0 + 32);

    v16bf af[4];
#pragma unroll
    for (int mt = 0; mt < 4; ++mt) {
      const __bf16* p = &As[cur][wm * 64 + mt * 16 + ln][0];
      af[mt] = cat8(*(const v8bf*)(p + kbA), *(const v8bf*)(p + 16 + kbA));
    }
#pragma unroll
    for (int nt = 0; nt < 4; ++nt) {
      const __bf16* p = &Bs[cur][wn * 64 + nt * 16 + ln][0];
      v16bf bfr = cat8(*(const v8bf*)(p + kbB), *(const v8bf*)(p + kbB + 16));
#pragma unroll
      for (int mt = 0; mt < 4; ++mt)
        acc[mt][nt] = __builtin_amdgcn_wmma_f32_16x16x32_bf16(
            false, af[mt], false, bfr, (short)0, acc[mt][nt], false, false);
    }
  }

#pragma unroll
  for (int mt = 0; mt < 4; ++mt)
#pragma unroll
    for (int nt = 0; nt < 4; ++nt) {
      int col = n0 + wn * 64 + nt * 16 + ln;
      float bv = BIAS ? bias[col] : 0.0f;
#pragma unroll
      for (int r = 0; r < 8; ++r) {
        int row = m0 + wm * 64 + mt * 16 + r + 8 * g;   // C-layout row mapping
        float vv = acc[mt][nt][r] + bv;
        if (RELU) vv = fmaxf(vv, 0.0f);
        if (OUTBF) Cb[(size_t)row * Ndim + col] = f2bf(vv);
        else       Cf[(size_t)row * Ndim + col] = vv;
      }
    }
  (void)Mdim;
}

// ---------------- flash attention (bf16 WMMA, fp32 accum) ----------------
// Grid (B*NH, T/128); 256 threads = 8 waves; wave handles a 16-row query tile.
// q,k: [B,T,NH,HD] strided by H. vt: per-head [HD,T]. y: [B,T,NH,HD].
__global__ __launch_bounds__(256) void k_attn(
    const __bf16* __restrict__ q, const __bf16* __restrict__ k,
    const __bf16* __restrict__ vt, __bf16* __restrict__ y)
{
  __shared__ __bf16 Ps[8][16][48];   // per-wave P staging (16 x 32, padded)

  const int tid  = threadIdx.x;
  const int lane = tid & 31;
  const int w    = tid >> 5;
  const int bh   = blockIdx.x;
  const int b    = bh >> 3, h = bh & 7;
  const int q0   = blockIdx.y * 128 + w * 16;

  const __bf16* qh = q  + (size_t)b * TDIM * HDIM + h * HEADD;
  const __bf16* kh = k  + (size_t)b * TDIM * HDIM + h * HEADD;
  const __bf16* vh = vt + (size_t)bh * HEADD * TDIM;

  const int ln  = lane & 15;
  const int g   = lane >> 4;
  const int kbA = g * 8;
  const int kbB = g * 8;

  // Q fragments (16 x 64 -> two 16x32 A-frags), loaded once
  v16bf qf[2];
  {
    const __bf16* qr = qh + (size_t)(q0 + ln) * HDIM;
#pragma unroll
    for (int j = 0; j < 2; ++j)
      qf[j] = cat8(*(const v8bf*)(qr + 32 * j + kbA),
                   *(const v8bf*)(qr + 32 * j + 16 + kbA));
  }

  v8f O[4] = {};
  float mrow[8], lrow[8];
#pragma unroll
  for (int r = 0; r < 8; ++r) { mrow[r] = -3.0e38f; lrow[r] = 0.0f; }

  const float scale = 0.044194173824159216f;   // 1/sqrt(512) (full H per reference)
  const int kend = q0 + 16;

  for (int kt = 0; kt < kend; kt += 32) {
    // S = Q K^T for 32 keys: 2 N-tiles x 2 K-chunks
    v8f S[2] = {};
#pragma unroll
    for (int nt = 0; nt < 2; ++nt) {
#pragma unroll
      for (int j = 0; j < 2; ++j) {
        const __bf16* kr = kh + (size_t)(kt + nt * 16 + ln) * HDIM + 32 * j;
        v16bf kf = cat8(*(const v8bf*)(kr + kbB), *(const v8bf*)(kr + kbB + 16));
        S[nt] = __builtin_amdgcn_wmma_f32_16x16x32_bf16(
            false, qf[j], false, kf, (short)0, S[nt], false, false);
      }
    }

    // scale + causal mask + online softmax; stage P (bf16) in LDS
#pragma unroll
    for (int r = 0; r < 8; ++r) {
      int qrow = q0 + r + 8 * g;
      float s0 = S[0][r] * scale;
      float s1 = S[1][r] * scale;
      if (kt + ln > qrow)      s0 = -3.0e38f;
      if (kt + 16 + ln > qrow) s1 = -3.0e38f;
      float mx   = rmax16(fmaxf(s0, s1));
      float mnew = fmaxf(mrow[r], mx);
      float corr = __expf(mrow[r] - mnew);
      mrow[r] = mnew;
      lrow[r] *= corr;
#pragma unroll
      for (int ot = 0; ot < 4; ++ot) O[ot][r] *= corr;
      float p0 = __expf(s0 - mnew);
      float p1 = __expf(s1 - mnew);
      lrow[r] += rsum16(p0 + p1);
      Ps[w][r + 8 * g][ln]      = f2bf(p0);
      Ps[w][r + 8 * g][16 + ln] = f2bf(p1);
    }

    // wave-local LDS ordering (waves diverge in trip count; no __syncthreads)
    asm volatile("s_wait_dscnt 0x0" ::: "memory");

    // P fragment (A, 16x32) and O += P * V via Vt (B-frags contiguous)
    v16bf pf;
    {
      const __bf16* p = &Ps[w][ln][0];
      pf = cat8(*(const v8bf*)(p + kbA), *(const v8bf*)(p + 16 + kbA));
    }
#pragma unroll
    for (int ot = 0; ot < 4; ++ot) {
      const __bf16* vr = vh + (size_t)(16 * ot + ln) * TDIM + kt + kbB;
      v16bf vf = cat8(*(const v8bf*)(vr), *(const v8bf*)(vr + 16));
      O[ot] = __builtin_amdgcn_wmma_f32_16x16x32_bf16(
          false, pf, false, vf, (short)0, O[ot], false, false);
    }
  }

  // y = O / l
#pragma unroll
  for (int r = 0; r < 8; ++r) {
    float inv = 1.0f / lrow[r];
    int row = q0 + r + 8 * g;
    __bf16* yr = y + ((size_t)b * TDIM + row) * HDIM + h * HEADD;
#pragma unroll
    for (int ot = 0; ot < 4; ++ot)
      yr[16 * ot + ln] = f2bf(O[ot][r] * inv);
  }
}

// ---------------- launcher ----------------
extern "C" void kernel_launch(void* const* d_in, const int* in_sizes, int n_in,
                              void* d_out, int out_size, void* d_ws, size_t ws_size,
                              hipStream_t stream) {
  (void)in_sizes; (void)n_in; (void)out_size; (void)ws_size;
  const int*   ixs  = (const int*)  d_in[0];
  const float* tok  = (const float*)d_in[1];
  const float* pos  = (const float*)d_in[2];
  const float* Wprj = (const float*)d_in[3];
  const float* Wq   = (const float*)d_in[4];
  const float* bq   = (const float*)d_in[5];
  const float* Wk   = (const float*)d_in[6];
  const float* bk   = (const float*)d_in[7];
  const float* Wv   = (const float*)d_in[8];
  const float* bv   = (const float*)d_in[9];
  const float* W1   = (const float*)d_in[10];
  const float* b1   = (const float*)d_in[11];
  const float* W2   = (const float*)d_in[12];
  const float* b2   = (const float*)d_in[13];
  float* out = (float*)d_out;

  // workspace carve-up (bf16). Total ~69 MB.
  __bf16* ws = (__bf16*)d_ws;
  size_t off = 0;
  auto nxt = [&](size_t n) { __bf16* p = ws + off; off += n; return p; };
  const size_t MH = (size_t)MROWS * HDIM;
  __bf16* xe    = nxt(MH);
  __bf16* Wprjb = nxt((size_t)HDIM * HDIM);
  __bf16* Wqb   = nxt((size_t)HDIM * HDIM);
  __bf16* Wkb   = nxt((size_t)HDIM * HDIM);
  __bf16* Wvb   = nxt((size_t)HDIM * HDIM);
  __bf16* W1b   = nxt((size_t)HDIM * HDIM);
  __bf16* W2b   = nxt((size_t)VDIM * HDIM);
  __bf16* xp    = nxt(MH);
  __bf16* qb    = nxt(MH);
  __bf16* kb    = nxt(MH);
  __bf16* vb    = nxt(MH);
  __bf16* vtb   = nxt(MH);
  __bf16* yb    = nxt(MH);
  __bf16* h1b   = nxt(MH);

  const int HH = HDIM * HDIM;
  k_f32_to_bf16<<<(HH + 255) / 256, 256, 0, stream>>>(Wprj, Wprjb, HH);
  k_f32_to_bf16<<<(HH + 255) / 256, 256, 0, stream>>>(Wq,   Wqb,   HH);
  k_f32_to_bf16<<<(HH + 255) / 256, 256, 0, stream>>>(Wk,   Wkb,   HH);
  k_f32_to_bf16<<<(HH + 255) / 256, 256, 0, stream>>>(Wv,   Wvb,   HH);
  k_f32_to_bf16<<<(HH + 255) / 256, 256, 0, stream>>>(W1,   W1b,   HH);
  k_f32_to_bf16<<<(VDIM * HDIM + 255) / 256, 256, 0, stream>>>(W2, W2b, VDIM * HDIM);

  k_embed<<<(int)(MH / 256), 256, 0, stream>>>(ixs, tok, pos, xe);

  dim3 gH(HDIM / 256, MROWS / 128);                  // (2, 32)
  k_gemm_nt<0, 0, 1><<<gH, 256, 0, stream>>>(xe, Wprjb, nullptr, nullptr, xp, MROWS, HDIM, HDIM);
  k_gemm_nt<0, 1, 1><<<gH, 256, 0, stream>>>(xp, Wqb, bq, nullptr, qb, MROWS, HDIM, HDIM);
  k_gemm_nt<0, 1, 1><<<gH, 256, 0, stream>>>(xp, Wkb, bk, nullptr, kb, MROWS, HDIM, HDIM);
  k_gemm_nt<0, 1, 1><<<gH, 256, 0, stream>>>(xp, Wvb, bv, nullptr, vb, MROWS, HDIM, HDIM);

  k_transpose_v<<<(int)(MH / 256), 256, 0, stream>>>(vb, vtb);

  k_attn<<<dim3(BDIM * NHEAD, TDIM / 128), 256, 0, stream>>>(qb, kb, vtb, yb);

  k_gemm_nt<1, 1, 1><<<gH, 256, 0, stream>>>(yb, W1b, b1, nullptr, h1b, MROWS, HDIM, HDIM);

  dim3 gV(VDIM / 256, MROWS / 128);                  // (125, 32)
  k_gemm_nt<1, 1, 0><<<gV, 256, 0, stream>>>(h1b, W2b, b2, out, nullptr, MROWS, VDIM, HDIM);
}
